// CrossAttention_80547816669358
// MI455X (gfx1250) — matrix operand order repaired and checked by reading, not compiled, not run on hardware
//
#include <hip/hip_runtime.h>

typedef __attribute__((ext_vector_type(16))) _Float16 v16h;
typedef __attribute__((ext_vector_type(8)))  _Float16 v8h;
typedef __attribute__((ext_vector_type(4)))  _Float16 v4h;
typedef __attribute__((ext_vector_type(8)))  float    v8f;
typedef __attribute__((ext_vector_type(4)))  float    v4f;

#define BB  8
#define TB  256
#define SB  4096
#define EB  512
#define KVE 768
#define NH  8
#define HD  64

#define WMMA_F16(A, B, C) \
    __builtin_amdgcn_wmma_f32_16x16x32_f16(false, (A), false, (B), (short)0, (C), false, false)

// ---- helpers -------------------------------------------------------------

// B-operand: 16 contiguous halves at p (caller folds 16*hh into p).
static __device__ __forceinline__ v16h loadB16(const _Float16* p) {
    v8h a = *(const v8h*)p;
    v8h b = *(const v8h*)(p + 8);
    return __builtin_shufflevector(a, b, 0,1,2,3,4,5,6,7,8,9,10,11,12,13,14,15);
}

// A-operand: 16 contiguous halves at p (caller folds 16*hh into p).
static __device__ __forceinline__ v16h loadA16(const _Float16* p) {
    v8h a = *(const v8h*)p;
    v8h b = *(const v8h*)(p + 8);
    return __builtin_shufflevector(a, b, 0,1,2,3,4,5,6,7,8,9,10,11,12,13,14,15);
}

// DPP cross-lane moves: reductions over a 16-lane row with no LDS traffic.
template<int CTRL>
static __device__ __forceinline__ float dpp_mov(float v) {
    return __builtin_bit_cast(float,
        __builtin_amdgcn_update_dpp(0, __builtin_bit_cast(int, v),
                                    CTRL, 0xf, 0xf, true));
}
static __device__ __forceinline__ float redmax16(float v) {
    v = fmaxf(v, dpp_mov<0x0B1>(v));   // quad_perm(1,0,3,2)  : xor 1
    v = fmaxf(v, dpp_mov<0x04E>(v));   // quad_perm(2,3,0,1)  : xor 2
    v = fmaxf(v, dpp_mov<0x141>(v));   // row_half_mirror     : xor within 8
    v = fmaxf(v, dpp_mov<0x140>(v));   // row_mirror          : full 16 lanes
    return v;
}
static __device__ __forceinline__ float redsum16(float v) {
    v += dpp_mov<0x0B1>(v);
    v += dpp_mov<0x04E>(v);
    v += dpp_mov<0x141>(v);
    v += dpp_mov<0x140>(v);
    return v;
}

// ---- f32 -> f16 convert (vectorized: b128 in, b64 out) -------------------

__global__ void __launch_bounds__(256)
cvt_f32_f16_x4(const v4f* __restrict__ in, v4h* __restrict__ out, int n4) {
    int i = blockIdx.x * blockDim.x + threadIdx.x;
    int stride = gridDim.x * blockDim.x;
    for (; i < n4; i += stride)
        out[i] = __builtin_convertvector(in[i], v4h);
}

// ---- generic WMMA GEMM: out[m][n] = sum_k X[m][k] * W[n][k] --------------
// X: MxKK row-major f16, W: NxKK row-major f16. Each wave -> 32(M) x 64(N).
// KK compile-time: fully unrolled 2-stage ping-pong pipeline, all loads at
// base + immediate offset, no double-buffer copies, no OOB loads.

template<int KK>
__global__ void __launch_bounds__(256)
gemm_wmma(const _Float16* __restrict__ X, const _Float16* __restrict__ W,
          _Float16* __restrict__ outh, float* __restrict__ outf,
          const float* __restrict__ bias, int M, int N, float scale)
{
    const int lane = threadIdx.x & 31;
    const int wave = threadIdx.x >> 5;
    const int l16  = lane & 15;
    const int hh   = lane >> 4;
    const int Mtiles = M >> 5, Ntiles = N >> 6;
    const int w = blockIdx.x * 8 + wave;
    if (w >= Mtiles * Ntiles) return;
    const int mt = w / Ntiles, nt = w % Ntiles;
    const int n0 = nt * 64;

    const _Float16* pA0 = X + (size_t)(mt*32 + l16) * KK + 16*hh;
    const _Float16* pA1 = pA0 + (size_t)16 * KK;
    const _Float16* pW0 = W + (size_t)(n0 +  0 + l16) * KK + 16*hh;
    const _Float16* pW1 = W + (size_t)(n0 + 16 + l16) * KK + 16*hh;
    const _Float16* pW2 = W + (size_t)(n0 + 32 + l16) * KK + 16*hh;
    const _Float16* pW3 = W + (size_t)(n0 + 48 + l16) * KK + 16*hh;

    v8f acc0[4] = {}, acc1[4] = {};

    // prologue: k = 0 operands
    v16h a0c = loadA16(pA0), a1c = loadA16(pA1);
    v16h b0c = loadB16(pW0), b1c = loadB16(pW1);
    v16h b2c = loadB16(pW2), b3c = loadB16(pW3);

    #pragma unroll
    for (int k0 = 0; k0 < KK - 64; k0 += 64) {
        // stage 1: load k0+32 while computing k0
        v16h a0n = loadA16(pA0 + k0 + 32), a1n = loadA16(pA1 + k0 + 32);
        v16h b0n = loadB16(pW0 + k0 + 32), b1n = loadB16(pW1 + k0 + 32);
        v16h b2n = loadB16(pW2 + k0 + 32), b3n = loadB16(pW3 + k0 + 32);
        acc0[0] = WMMA_F16(a0c, b0c, acc0[0]);  acc1[0] = WMMA_F16(a1c, b0c, acc1[0]);
        acc0[1] = WMMA_F16(a0c, b1c, acc0[1]);  acc1[1] = WMMA_F16(a1c, b1c, acc1[1]);
        acc0[2] = WMMA_F16(a0c, b2c, acc0[2]);  acc1[2] = WMMA_F16(a1c, b2c, acc1[2]);
        acc0[3] = WMMA_F16(a0c, b3c, acc0[3]);  acc1[3] = WMMA_F16(a1c, b3c, acc1[3]);
        // stage 2: load k0+64 while computing k0+32
        a0c = loadA16(pA0 + k0 + 64);  a1c = loadA16(pA1 + k0 + 64);
        b0c = loadB16(pW0 + k0 + 64);  b1c = loadB16(pW1 + k0 + 64);
        b2c = loadB16(pW2 + k0 + 64);  b3c = loadB16(pW3 + k0 + 64);
        acc0[0] = WMMA_F16(a0n, b0n, acc0[0]);  acc1[0] = WMMA_F16(a1n, b0n, acc1[0]);
        acc0[1] = WMMA_F16(a0n, b1n, acc0[1]);  acc1[1] = WMMA_F16(a1n, b1n, acc1[1]);
        acc0[2] = WMMA_F16(a0n, b2n, acc0[2]);  acc1[2] = WMMA_F16(a1n, b2n, acc1[2]);
        acc0[3] = WMMA_F16(a0n, b3n, acc0[3]);  acc1[3] = WMMA_F16(a1n, b3n, acc1[3]);
    }

    // epilogue: last 64 (k = KK-64 operands already in c-buffers)
    {
        v16h a0n = loadA16(pA0 + KK - 32), a1n = loadA16(pA1 + KK - 32);
        v16h b0n = loadB16(pW0 + KK - 32), b1n = loadB16(pW1 + KK - 32);
        v16h b2n = loadB16(pW2 + KK - 32), b3n = loadB16(pW3 + KK - 32);
        acc0[0] = WMMA_F16(a0c, b0c, acc0[0]);  acc1[0] = WMMA_F16(a1c, b0c, acc1[0]);
        acc0[1] = WMMA_F16(a0c, b1c, acc0[1]);  acc1[1] = WMMA_F16(a1c, b1c, acc1[1]);
        acc0[2] = WMMA_F16(a0c, b2c, acc0[2]);  acc1[2] = WMMA_F16(a1c, b2c, acc1[2]);
        acc0[3] = WMMA_F16(a0c, b3c, acc0[3]);  acc1[3] = WMMA_F16(a1c, b3c, acc1[3]);
        acc0[0] = WMMA_F16(a0n, b0n, acc0[0]);  acc1[0] = WMMA_F16(a1n, b0n, acc1[0]);
        acc0[1] = WMMA_F16(a0n, b1n, acc0[1]);  acc1[1] = WMMA_F16(a1n, b1n, acc1[1]);
        acc0[2] = WMMA_F16(a0n, b2n, acc0[2]);  acc1[2] = WMMA_F16(a1n, b2n, acc1[2]);
        acc0[3] = WMMA_F16(a0n, b3n, acc0[3]);  acc1[3] = WMMA_F16(a1n, b3n, acc1[3]);
    }

    #pragma unroll
    for (int st = 0; st < 4; ++st) {
        const int n = n0 + st*16 + l16;
        #pragma unroll
        for (int r = 0; r < 8; ++r) {
            const int r0 = mt*32 + r + 8*hh;
            const float v0 = acc0[st][r] * scale;
            const float v1 = acc1[st][r] * scale;
            if (outf) {
                const float bv = bias ? bias[n] : 0.0f;
                outf[(size_t)r0 * N + n]        = v0 + bv;
                outf[(size_t)(r0 + 16) * N + n] = v1 + bv;
            }
            if (outh) {
                outh[(size_t)r0 * N + n]        = (_Float16)v0;
                outh[(size_t)(r0 + 16) * N + n] = (_Float16)v1;
            }
        }
    }
}

// ---- repack V from KV (b,s,{k|v},h,d) into V^T (b,h,d,s) -----------------
// Stage the 64x64 tile into LDS with the CDNA5 async global->LDS pipe
// (no VGPR round trip, tracked by ASYNCcnt), then do the bank-conflict-free
// transposed read (stride 66 halves = 132 B -> bank = 33*col mod 64).

__global__ void __launch_bounds__(256)
repack_vt(const _Float16* __restrict__ KV, _Float16* __restrict__ VT)
{
    __shared__ _Float16 tr[64][66];      // [s][d], +2 halves pad (4B-aligned rows)
    const int blk = blockIdx.x;          // B*NH*(SB/64) = 4096
    const int sc = blk & 63;
    const int h  = (blk >> 6) & 7;
    const int b  = blk >> 9;
    const int s0 = sc * 64;
    const int tid = threadIdx.x;

    // phase 1: async-load 64 rows x 128 B as 2048 b32 chunks (8 per thread)
    #pragma unroll
    for (int r = 0; r < 8; ++r) {
        const int idx   = r * 256 + tid;      // 0..2047
        const int srow  = idx >> 5;           // s within tile
        const int cpair = idx & 31;           // d pair
        const _Float16* gp = KV + (size_t)(b*SB + s0 + srow) * 1024
                                + 512 + h*HD + cpair*2;
        unsigned lds_addr = (unsigned)(size_t)&tr[srow][cpair*2];
        asm volatile("global_load_async_to_lds_b32 %0, %1, off"
                     :: "v"(lds_addr), "v"((unsigned long long)(size_t)gp));
    }
    asm volatile("s_wait_asynccnt 0" ::: "memory");
    __syncthreads();

    // phase 2: coalesced transposed stores
    const int col   = tid & 63;          // s
    const int rbase = tid >> 6;          // 0..3
    #pragma unroll
    for (int r = 0; r < 16; ++r) {
        const int d = r*4 + rbase;
        VT[(size_t)((b*NH + h)*HD + d) * SB + s0 + col] = tr[col][d];
    }
}

// ---- flash-attention: one wave per (b, h, 16-row t-tile) -----------------
// Q (b,t,h*64+d) pre-scaled f16; K in KV at (b,s,0,h,d); V^T (b,h,d,s).
// K tiles double-buffered across s-steps (unroll 2 renames the ping-pong);
// V tiles loaded before the softmax; softmax reductions are pure-VALU DPP
// butterflies (no LDS), so the only LDS use is the P-tile transpose.

__global__ void __launch_bounds__(256)
attn_wmma(const _Float16* __restrict__ Q, const _Float16* __restrict__ KVh,
          const _Float16* __restrict__ VT, const int* __restrict__ mask,
          _Float16* __restrict__ AO)
{
    __shared__ _Float16 plds[8 * 16 * 32];   // per-wave 16x32 P tile
    const int lane = threadIdx.x & 31;
    const int wave = threadIdx.x >> 5;
    const int l16  = lane & 15;
    const int hh   = lane >> 4;
    const int w = blockIdx.x * 8 + wave;     // 0..1023
    const int b  = w >> 7;
    const int h  = (w >> 4) & 7;
    const int t0 = (w & 15) * 16;

    _Float16* P = plds + wave * 512;

    const _Float16* qrow = Q + (size_t)(b*TB + t0 + l16) * EB + h*HD + 16*hh;
    const v16h qa0 = loadA16(qrow);        // d 0..31
    const v16h qa1 = loadA16(qrow + 32);   // d 32..63

    v8f Oacc[4] = {};
    float mrow[8], lrow[8];
    #pragma unroll
    for (int r = 0; r < 8; ++r) { mrow[r] = -1e30f; lrow[r] = 0.0f; }

    const int*      mptr  = mask + b*SB;
    // per-lane K row pointer (s = l16) and V^T row pointer (d = l16)
    const _Float16* Kbase = KVh + (size_t)(b*SB + l16) * 1024 + h*HD + 16*hh;
    const _Float16* Vbase = VT  + (size_t)((b*NH + h)*HD + l16) * SB + 16*hh;

    // prologue: K tiles for s0 = 0
    v16h kb00 = loadB16(Kbase);
    v16h kb01 = loadB16(Kbase + 32);
    v16h kb10 = loadB16(Kbase + (size_t)16*1024);
    v16h kb11 = loadB16(Kbase + (size_t)16*1024 + 32);

    #pragma unroll 2
    for (int s0 = 0; s0 < SB; s0 += 32) {
        // ---- scores from pre-loaded K tiles
        v8f sc0 = {}, sc1 = {};
        sc0 = WMMA_F16(qa0, kb00, sc0);
        sc0 = WMMA_F16(qa1, kb01, sc0);
        sc1 = WMMA_F16(qa0, kb10, sc1);
        sc1 = WMMA_F16(qa1, kb11, sc1);

        // ---- issue next-step K loads immediately (hide under softmax)
        const int sn = (s0 + 32 < SB) ? s0 + 32 : 0;   // wrap: in-bounds dummy on tail
        const _Float16* kp = Kbase + (size_t)sn * 1024;
        const v16h nk00 = loadB16(kp);
        const v16h nk01 = loadB16(kp + 32);
        const v16h nk10 = loadB16(kp + (size_t)16*1024);
        const v16h nk11 = loadB16(kp + (size_t)16*1024 + 32);

        // ---- V tiles for the current step (consumed after LDS round trip)
        v16h vb[4];
        #pragma unroll
        for (int dt = 0; dt < 4; ++dt)
            vb[dt] = loadB16(Vbase + (size_t)dt*16*SB + s0);

        if (s0 + 64 < SB)   // keep the stripe after next warm in L2
            __builtin_prefetch(Kbase + (size_t)(s0 + 64) * 1024, 0, 1);

        const int mk0 = mptr[s0 + l16];
        const int mk1 = mptr[s0 + 16 + l16];
        #pragma unroll
        for (int r = 0; r < 8; ++r) {
            if (mk0) sc0[r] = -1e30f;
            if (mk1) sc1[r] = -1e30f;
        }

        // ---- row max over 32 cols: DPP butterfly per 16-lane row
        float tm[8], al[8];
        #pragma unroll
        for (int r = 0; r < 8; ++r)
            tm[r] = redmax16(fmaxf(sc0[r], sc1[r]));
        #pragma unroll
        for (int r = 0; r < 8; ++r) {
            const float mn = fmaxf(mrow[r], tm[r]);
            al[r] = __expf(mrow[r] - mn);
            mrow[r] = mn;
        }

        float p0[8], p1[8], rs[8];
        #pragma unroll
        for (int r = 0; r < 8; ++r) {
            p0[r] = mk0 ? 0.0f : __expf(sc0[r] - mrow[r]);
            p1[r] = mk1 ? 0.0f : __expf(sc1[r] - mrow[r]);
            rs[r] = redsum16(p0[r] + p1[r]);
        }
        #pragma unroll
        for (int r = 0; r < 8; ++r) {
            lrow[r] = lrow[r] * al[r] + rs[r];
            #pragma unroll
            for (int dt = 0; dt < 4; ++dt) Oacc[dt][r] *= al[r];
        }

        // ---- C-layout -> A-layout transpose of P through LDS
        // (LDS ops are in-order within a wave; wave_barrier only stops the
        //  compiler from reordering across the cross-lane exchange.)
        #pragma unroll
        for (int r = 0; r < 8; ++r) {
            P[(r + 8*hh)*32 + l16]      = (_Float16)p0[r];
            P[(r + 8*hh)*32 + 16 + l16] = (_Float16)p1[r];
        }
        __builtin_amdgcn_wave_barrier();
        const v16h pa = loadA16(P + l16*32 + 16*hh);

        // ---- O += P @ V (4 d-tiles of 16)
        #pragma unroll
        for (int dt = 0; dt < 4; ++dt)
            Oacc[dt] = WMMA_F16(pa, vb[dt], Oacc[dt]);
        __builtin_amdgcn_wave_barrier();

        kb00 = nk00; kb01 = nk01; kb10 = nk10; kb11 = nk11;
    }

    // ---- normalize and store f16 row-major (b, t, h*64+d)
    float inv[8];
    #pragma unroll
    for (int r = 0; r < 8; ++r) inv[r] = 1.0f / lrow[r];
    _Float16* aorow = AO + (size_t)(b*TB + t0) * EB + h*HD;
    #pragma unroll
    for (int dt = 0; dt < 4; ++dt) {
        #pragma unroll
        for (int r = 0; r < 8; ++r)
            aorow[(size_t)(r + 8*hh) * EB + dt*16 + l16] = (_Float16)(Oacc[dt][r] * inv[r]);
    }
}

// ---- host-side orchestration ---------------------------------------------

extern "C" void kernel_launch(void* const* d_in, const int* in_sizes, int n_in,
                              void* d_out, int out_size, void* d_ws, size_t ws_size,
                              hipStream_t stream) {
    (void)in_sizes; (void)n_in; (void)out_size; (void)ws_size;
    const float* x    = (const float*)d_in[0];   // (B,T,E)
    const float* ctx  = (const float*)d_in[1];   // (B,S,KV)
    const int*   mask = (const int*)d_in[2];     // (B,S)
    const float* Wq   = (const float*)d_in[3];   // (512,512)
    const float* Wkv  = (const float*)d_in[4];   // (1024,768)
    const float* Wo   = (const float*)d_in[5];   // (512,512)
    const float* bo   = (const float*)d_in[6];   // (512,)
    float* out = (float*)d_out;

    char* ws = (char*)d_ws;
    _Float16* x_h   = (_Float16*)ws; ws += (size_t)BB*TB*EB * 2;        // 2 MB
    _Float16* Wq_h  = (_Float16*)ws; ws += (size_t)EB*EB * 2;           // 0.5 MB
    _Float16* Wkv_h = (_Float16*)ws; ws += (size_t)1024*KVE * 2;        // 1.5 MB
    _Float16* Wo_h  = (_Float16*)ws; ws += (size_t)EB*EB * 2;           // 0.5 MB
    _Float16* ctx_h = (_Float16*)ws; ws += (size_t)BB*SB*KVE * 2;       // 48 MB
    _Float16* Q_h   = (_Float16*)ws; ws += (size_t)BB*TB*EB * 2;        // 2 MB
    _Float16* KV_h  = (_Float16*)ws; ws += (size_t)BB*SB*1024 * 2;      // 64 MB
    _Float16* VT_h  = (_Float16*)ws; ws += (size_t)BB*NH*HD*SB * 2;     // 32 MB
    _Float16* AO_h  = (_Float16*)ws; ws += (size_t)BB*TB*EB * 2;        // 2 MB

    // 1) converts (x4 vectorized; all element counts divisible by 4)
    cvt_f32_f16_x4<<<512,  256, 0, stream>>>((const v4f*)x,   (v4h*)x_h,   BB*TB*EB/4);
    cvt_f32_f16_x4<<<128,  256, 0, stream>>>((const v4f*)Wq,  (v4h*)Wq_h,  EB*EB/4);
    cvt_f32_f16_x4<<<128,  256, 0, stream>>>((const v4f*)Wkv, (v4h*)Wkv_h, 1024*KVE/4);
    cvt_f32_f16_x4<<<128,  256, 0, stream>>>((const v4f*)Wo,  (v4h*)Wo_h,  EB*EB/4);
    cvt_f32_f16_x4<<<4096, 256, 0, stream>>>((const v4f*)ctx, (v4h*)ctx_h, BB*SB*KVE/4);

    // 2) Q = x @ Wq^T, scaled by 1/sqrt(D), stored f16
    //    waves = (2048/32)*(512/64) = 512 -> 64 blocks
    gemm_wmma<512><<<64, 256, 0, stream>>>(x_h, Wq_h, Q_h, nullptr, nullptr,
                                           BB*TB, EB, 0.125f);

    // 3) KV = context @ Wkv^T : waves = (32768/32)*(1024/64) = 16384 -> 2048 blocks
    gemm_wmma<768><<<2048, 256, 0, stream>>>(ctx_h, Wkv_h, KV_h, nullptr, nullptr,
                                             BB*SB, 1024, 1.0f);

    // 4) V -> V^T (b,h,d,s) via async global->LDS staging
    repack_vt<<<4096, 256, 0, stream>>>(KV_h, VT_h);

    // 5) flash attention: 1024 waves -> 128 blocks
    attn_wmma<<<128, 256, 0, stream>>>(Q_h, KV_h, VT_h, mask, AO_h);

    // 6) out = AO @ Wo^T + bo (f32)
    gemm_wmma<512><<<64, 256, 0, stream>>>(AO_h, Wo_h, nullptr, out, bo,
                                           BB*TB, EB, 1.0f);
}
